// StructureDecoder_9663676416842
// MI455X (gfx1250) — matrix operand order repaired and checked by reading, not compiled, not run on hardware
//
#include <hip/hip_runtime.h>
#include <stdint.h>

typedef __attribute__((ext_vector_type(2)))  float    v2f;
typedef __attribute__((ext_vector_type(8)))  float    v8f;
typedef __attribute__((ext_vector_type(8)))  _Float16 v8h;
typedef __attribute__((ext_vector_type(16))) _Float16 v16h;

#define NN    8192
#define NH    128
#define NE    262144
#define KK    64
#define XKSTR 132   // padded LDS stride (floats) to avoid 64-bank conflicts; 528B rows stay 16B-aligned
#define GSTR  68

// ---------------- adjacency build ----------------

__global__ void init_kernel(int* rowcount, int* colcount, int* rowfill) {
    int i = blockIdx.x * blockDim.x + threadIdx.x;
    if (i < NN) { rowcount[i] = 0; colcount[i] = 1; rowfill[i] = 0; } // colcount=1: self loop
}

__global__ void count_kernel(const int* __restrict__ er, const int* __restrict__ ec,
                             int* rowcount, int* colcount) {
    int e = blockIdx.x * blockDim.x + threadIdx.x;
    if (e < NE) {
        atomicAdd(&rowcount[er[e]], 1);
        atomicAdd(&colcount[ec[e]], 1);
    }
}

__global__ __launch_bounds__(256) void scan_kernel(const int* __restrict__ rowcount, int* rowptr) {
    __shared__ int partial[256];
    __shared__ int base[256];
    const int tid = threadIdx.x;
    const int start = tid * 32;
    int s = 0;
    for (int j = 0; j < 32; ++j) s += rowcount[start + j];
    partial[tid] = s;
    __syncthreads();
    if (tid == 0) {
        int acc = 0;
        for (int t = 0; t < 256; ++t) { base[t] = acc; acc += partial[t]; }
        rowptr[NN] = acc;
    }
    __syncthreads();
    int acc = base[tid];
    for (int j = 0; j < 32; ++j) { rowptr[start + j] = acc; acc += rowcount[start + j]; }
}

__global__ void dinv_kernel(const int* __restrict__ colcount, float* dinv) {
    int i = blockIdx.x * blockDim.x + threadIdx.x;
    if (i < NN) dinv[i] = rsqrtf((float)colcount[i]);
}

__global__ void fill_kernel(const int* __restrict__ er, const int* __restrict__ ec,
                            const int* __restrict__ rowptr, int* rowfill, int* csr) {
    int e = blockIdx.x * blockDim.x + threadIdx.x;
    if (e < NE) {
        int r = er[e];
        int pos = atomicAdd(&rowfill[r], 1);
        csr[rowptr[r] + pos] = ec[e];
    }
}

// ---------------- h = x @ W  (fp32 WMMA 16x16x4) ----------------
// block = 256 threads (8 waves); block b covers rows [16b,16b+16); wave w covers cols [16w,16w+16)
__global__ __launch_bounds__(256) void gemm_h_kernel(const float* __restrict__ x,
                                                     const float* __restrict__ W,
                                                     float* __restrict__ h) {
    const int wave = threadIdx.x >> 5;
    const int lane = threadIdx.x & 31;
    const int lh = lane & 15, lg = lane >> 4;
    const int rowbase = blockIdx.x * 16;
    const int colbase = wave * 16;
    v8f acc = {0.f,0.f,0.f,0.f,0.f,0.f,0.f,0.f};
    const float* xr = x + (rowbase + lh) * NH;
    for (int k0 = 0; k0 < NH; k0 += 4) {
        const int ka = k0 + 2 * lg;
        v2f A = { xr[ka], xr[ka + 1] };
        v2f B = { W[ka * NH + colbase + lh], W[(ka + 1) * NH + colbase + lh] };
        acc = __builtin_amdgcn_wmma_f32_16x16x4_f32(false, A, false, B, (short)0, acc, false, false);
    }
#pragma unroll
    for (int r = 0; r < 8; ++r)
        h[(rowbase + r + 8 * lg) * NH + colbase + lh] = acc[r];
}

// ---------------- per-node coalesce + top-64 ----------------
// one thread per node, LDS-resident candidate list
__global__ __launch_bounds__(64) void topk_kernel(const int* __restrict__ csr,
                                                  const int* __restrict__ rowptr,
                                                  const float* __restrict__ dinv,
                                                  int* __restrict__ tki, float* __restrict__ tkw,
                                                  float* __restrict__ rowsum) {
    __shared__ float swv[64 * KK];
    __shared__ int   sid[64 * KK];
    const int lt = threadIdx.x;
    const int node = blockIdx.x * 64 + lt;
    float* w = &swv[lt * KK];
    int*  id = &sid[lt * KK];
    const float dv = dinv[node];
    w[0] = dv * dv; id[0] = node;           // injected self loop
    int cnt = 1;
    float rs = dv * dv;
    const int s = rowptr[node], e = rowptr[node + 1];
    for (int p = s; p < e; ++p) {
        const int c = csr[p];
        const float w1 = dv * dinv[c];
        rs += w1;                            // row_sum counts every edge occurrence
        bool dup = false;
        for (int q = s; q < p; ++q) if (csr[q] == c) { dup = true; break; }
        if (dup) continue;
        int m = 1;
        for (int q = p + 1; q < e; ++q) if (csr[q] == c) ++m;
        const float wt = (float)m * w1;      // coalesced weight
        if (c == node) { w[0] += wt; continue; }
        if (cnt < KK) { id[cnt] = c; w[cnt] = wt; ++cnt; }
        else {
            int mi = 0; float mw = w[0];
            for (int q = 1; q < KK; ++q) if (w[q] < mw) { mw = w[q]; mi = q; }
            if (wt > mw) { w[mi] = wt; id[mi] = c; }
        }
    }
    rowsum[node] = rs;
    for (int q = cnt; q < KK; ++q) { w[q] = 0.f; id[q] = 0; }
    for (int q = 0; q < KK; ++q) { tkw[node * KK + q] = w[q]; tki[node * KK + q] = id[q]; }
}

// ---------------- soft k-medoid per node ----------------
// block = 256 threads (8 waves), one node per block. Neighbor gather goes straight to LDS
// via the CDNA5 async path (ASYNCcnt), then fp32 WMMA Gram on the 64x128 neighborhood.
__global__ __launch_bounds__(256) void medoid_kernel(const float* __restrict__ h,
                                                     const int* __restrict__ tki,
                                                     const float* __restrict__ tkw,
                                                     const float* __restrict__ rowsum,
                                                     const float* __restrict__ bias,
                                                     _Float16* __restrict__ Y) {
    __shared__ float xk[KK * XKSTR];
    __shared__ float G[KK * GSTR];
    __shared__ float sw[KK];
    __shared__ int   sidx[KK];
    __shared__ float sq[KK];
    __shared__ float part[KK * 4];
    __shared__ float lv[KK];
    __shared__ float cc[KK];
    __shared__ float sscale;

    const int node = blockIdx.x;
    const int tid  = threadIdx.x;
    if (tid < KK) { sw[tid] = tkw[node * KK + tid]; sidx[tid] = tki[node * KK + tid]; }
    __syncthreads();

    // async gather x_k = h[tk_idx] (64 x 128 fp32, 32KB) directly into LDS:
    // 2048 B128 chunks, 8 per thread. global_load_async_to_lds_b128 bypasses VGPRs.
    {
        const unsigned ldsbase = (unsigned)(uintptr_t)(&xk[0]);   // LDS aperture: offset in addr[31:0]
#pragma unroll
        for (int i = 0; i < 8; ++i) {
            const int c  = tid + 256 * i;                 // chunk id
            const int j  = c >> 5;                        // neighbor row 0..63
            const int dd = (c & 31) * 4;                  // float offset within row
            const float* gp = h + (size_t)sidx[j] * NH + dd;
            const unsigned lo = ldsbase + (unsigned)((j * XKSTR + dd) * 4);
            asm volatile("global_load_async_to_lds_b128 %0, %1, off"
                         :: "v"(lo), "v"(gp) : "memory");
        }
        asm volatile("s_wait_asynccnt 0x0" ::: "memory");
    }
    __syncthreads();

    // G = x_k x_k^T via V_WMMA_F32_16X16X4_F32; 16 tiles of 16x16, 2 per wave
    const int wave = tid >> 5;
    const int lane = tid & 31;
    const int lh = lane & 15, lg = lane >> 4;
    for (int t = wave; t < 16; t += 8) {
        const int ti = t >> 2, tj = t & 3;
        v8f acc = {0.f,0.f,0.f,0.f,0.f,0.f,0.f,0.f};
        for (int k0 = 0; k0 < NH; k0 += 4) {
            const int ka = k0 + 2 * lg;
            v2f A = { xk[(ti * 16 + lh) * XKSTR + ka], xk[(ti * 16 + lh) * XKSTR + ka + 1] };
            v2f B = { xk[(tj * 16 + lh) * XKSTR + ka], xk[(tj * 16 + lh) * XKSTR + ka + 1] };
            acc = __builtin_amdgcn_wmma_f32_16x16x4_f32(false, A, false, B, (short)0, acc, false, false);
        }
#pragma unroll
        for (int r = 0; r < 8; ++r)
            G[(ti * 16 + r + 8 * lg) * GSTR + tj * 16 + lh] = acc[r];
    }
    __syncthreads();

    if (tid < KK) sq[tid] = G[tid * GSTR + tid];
    __syncthreads();

    // l[j] = sum_m w[m] * dist(j,m)   (each group of 4 threads owns one j)
    {
        const int j = tid >> 2, q = tid & 3;
        float p = 0.f;
        for (int m = q * 16; m < q * 16 + 16; ++m) {
            const float d2 = sq[j] + sq[m] - 2.f * G[j * GSTR + m];
            p += sw[m] * sqrtf(fmaxf(d2, 1e-12f));
        }
        part[j * 4 + q] = p;
    }
    __syncthreads();
    if (tid < KK) {
        const float L = part[tid * 4] + part[tid * 4 + 1] + part[tid * 4 + 2] + part[tid * 4 + 3];
        lv[tid] = (sw[tid] > 0.f) ? L : 3.0e38f;
    }
    __syncthreads();

    // softmax + weight correction (serial over 64 in one lane; negligible)
    if (tid == 0) {
        const float rs = rowsum[node];
        const float T = rs;                  // temperature == 1.0
        float mn = 3.0e38f;
        for (int j = 0; j < KK; ++j) mn = fminf(mn, lv[j]);
        float csum = 0.f;
        for (int j = 0; j < KK; ++j) {
            const float e = (sw[j] > 0.f) ? __expf(-(lv[j] - mn) / T) : 0.f;
            const float cj = e * sw[j];
            cc[j] = cj; csum += cj;
        }
        sscale = rs / csum;
    }
    __syncthreads();

    // out[d] = relu(row_sum * sum_j c[j]*xk[j][d] + b[d]) -> f16 for the final Gram
    if (tid < NH) {
        float acc = 0.f;
        for (int j = 0; j < KK; ++j) acc += cc[j] * xk[j * XKSTR + tid];
        float val = acc * sscale + bias[tid];
        val = fmaxf(val, 0.f);
        Y[(size_t)node * NH + tid] = (_Float16)val;
    }
}

// ---------------- final structure Gram: out @ out^T (f16 WMMA, store-bound) ----------------
// block = 256 threads (8 waves); 128x128 output tile per block; Y (2MB) is L2-resident.
__global__ __launch_bounds__(256) void gram_kernel(const _Float16* __restrict__ Y,
                                                   float* __restrict__ out) {
    const int wave = threadIdx.x >> 5;
    const int lane = threadIdx.x & 31;
    const int lh = lane & 15, lg = lane >> 4;
    const int rowbase = blockIdx.y * 128 + wave * 16;
    const int colblk  = blockIdx.x * 128;

    // A fragments: contiguous k order chosen here (k = 16*lg + s), same as B -> lane<16: K[0..15]; lane>=16: K[16..31]
    v16h afrag[4];
    const _Float16* arow = Y + (size_t)(rowbase + lh) * NH;
#pragma unroll
    for (int kc = 0; kc < 4; ++kc) {
        const int kb = kc * 32;
        const v8h lo = *(const v8h*)(arow + kb + lg * 16);
        const v8h hi = *(const v8h*)(arow + kb + lg * 16 + 8);
#pragma unroll
        for (int i = 0; i < 8; ++i) { afrag[kc][i] = lo[i]; afrag[kc][8 + i] = hi[i]; }
    }

#pragma unroll
    for (int cs = 0; cs < 8; ++cs) {
        const int colbase = colblk + cs * 16;
        // B fragments: contiguous k order matching A -> lane<16: K[0..15] of column N=lh; lane>=16: K[16..31]
        const _Float16* brow = Y + (size_t)(colbase + lh) * NH + lg * 16;
        v8f acc = {0.f,0.f,0.f,0.f,0.f,0.f,0.f,0.f};
#pragma unroll
        for (int kc = 0; kc < 4; ++kc) {
            v16h bfrag = *(const v16h*)(brow + kc * 32);
            acc = __builtin_amdgcn_wmma_f32_16x16x32_f16(false, afrag[kc], false, bfrag,
                                                         (short)0, acc, false, false);
        }
#pragma unroll
        for (int r = 0; r < 8; ++r)
            out[(size_t)(rowbase + r + 8 * lg) * NN + colbase + lh] = acc[r];
    }
}

// ---------------- launch ----------------

extern "C" void kernel_launch(void* const* d_in, const int* in_sizes, int n_in,
                              void* d_out, int out_size, void* d_ws, size_t ws_size,
                              hipStream_t stream) {
    const float* x  = (const float*)d_in[0];
    const int*   ei = (const int*)d_in[1];     // [2, E]: rows then cols
    const float* W  = (const float*)d_in[2];
    const float* b  = (const float*)d_in[3];
    float* out = (float*)d_out;
    const int* er = ei;
    const int* ec = ei + NE;

    char* p = (char*)d_ws;
    auto alloc = [&](size_t bytes) -> void* {
        void* r = (void*)p;
        p += (bytes + 255) & ~(size_t)255;
        return r;
    };
    float*    h        = (float*)alloc((size_t)NN * NH * 4);
    _Float16* Y        = (_Float16*)alloc((size_t)NN * NH * 2);
    int*      rowcount = (int*)alloc((size_t)NN * 4);
    int*      colcount = (int*)alloc((size_t)NN * 4);
    int*      rowptr   = (int*)alloc((size_t)(NN + 1) * 4);
    int*      rowfill  = (int*)alloc((size_t)NN * 4);
    float*    dinv     = (float*)alloc((size_t)NN * 4);
    int*      csr      = (int*)alloc((size_t)NE * 4);
    int*      tki      = (int*)alloc((size_t)NN * KK * 4);
    float*    tkw      = (float*)alloc((size_t)NN * KK * 4);
    float*    rowsum   = (float*)alloc((size_t)NN * 4);

    init_kernel<<<NN / 256, 256, 0, stream>>>(rowcount, colcount, rowfill);
    count_kernel<<<NE / 256, 256, 0, stream>>>(er, ec, rowcount, colcount);
    scan_kernel<<<1, 256, 0, stream>>>(rowcount, rowptr);
    dinv_kernel<<<NN / 256, 256, 0, stream>>>(colcount, dinv);
    fill_kernel<<<NE / 256, 256, 0, stream>>>(er, ec, rowptr, rowfill, csr);
    gemm_h_kernel<<<NN / 16, 256, 0, stream>>>(x, W, h);
    topk_kernel<<<NN / 64, 64, 0, stream>>>(csr, rowptr, dinv, tki, tkw, rowsum);
    medoid_kernel<<<NN, 256, 0, stream>>>(h, tki, tkw, rowsum, b, Y);
    gram_kernel<<<dim3(NN / 128, NN / 128), 256, 0, stream>>>(Y, out);
}
